// GoogleAttention_59407987638594
// MI455X (gfx1250) — matrix operand order repaired and checked by reading, not compiled, not run on hardware
//
#include <hip/hip_runtime.h>

// ---------------- CDNA5 WMMA types ----------------
typedef __attribute__((ext_vector_type(16))) __bf16 v16bf;
typedef __attribute__((ext_vector_type(8)))  __bf16 v8bf;
typedef __attribute__((ext_vector_type(8)))  float  v8f;

union V16U { v16bf v; v8bf h[2]; };

__device__ __forceinline__ v8f wmma_bf16(v16bf a, v16bf b, v8f c) {
  // D = A(16x32 bf16) * B(32x16 bf16) + C(16x16 f32)
  return __builtin_amdgcn_wmma_f32_16x16x32_bf16(false, a, false, b, (short)0, c,
                                                 false, false);
}

// A operand: matrix stored row-major (rows = M, K contiguous), base pre-offset to (row0,k0).
// Lane L<16: M=L, K = 0..7,16..23 ; Lane L>=16: M=L-16, K = 8..15,24..31 (ISA 7.12.2).
__device__ __forceinline__ v16bf load_amat(const __bf16* base, int ld, int lane) {
  int m = lane & 15, half = lane >> 4;
  const __bf16* p = base + (size_t)m * ld + half * 8;
  V16U u;
  u.h[0] = *(const v8bf*)(p);        // K = kb .. kb+7
  u.h[1] = *(const v8bf*)(p + 16);   // K = 16+kb .. 16+kb+7
  return u.v;
}

// B operand: B[k][n] = M[n][k] with M stored row-major (rows = n, K contiguous).
// Lane L<16: n=L, K=0..7,16..23 ; Lane L>=16: n=L-16, K=8..15,24..31.
__device__ __forceinline__ v16bf load_bmat(const __bf16* base, int ld, int lane) {
  int ncol = lane & 15, kb = (lane >> 4) * 8;
  const __bf16* p = base + (size_t)ncol * ld + kb;
  V16U u;
  u.h[0] = *(const v8bf*)(p);
  u.h[1] = *(const v8bf*)(p + 16);
  return u.v;
}

// ---------------- problem constants ----------------
constexpr int kN   = 4;
constexpr int kC   = 256;
constexpr int kS   = 2304;          // 48*48
constexpr int kNH  = 8;
constexpr int kNT  = kS / 16;       // 144 tiles of 16 tokens
constexpr int kLPAD = 2312;         // f32 logits row stride (bank-skewed)
constexpr int kPPAD = 2320;         // bf16 prob row stride (16B aligned, bank-skewed)

// ---------------- prep: fp32 -> bf16 weights ----------------
__global__ void wconv_kernel(const float* __restrict__ wq, const float* __restrict__ wp,
                             __bf16* __restrict__ wqb, __bf16* __restrict__ wpb) {
  int i = blockIdx.x * 256 + threadIdx.x;
  if (i < 768 * 256) wqb[i] = (__bf16)wq[i];
  if (i < 256 * 256) wpb[i] = (__bf16)wp[i];
}

// ---------------- prep: x [N][C][S] f32 -> xT [N][S][C] bf16 (LDS tile transpose) ----------------
__global__ void xpose_kernel(const float* __restrict__ x, __bf16* __restrict__ xT) {
  __shared__ float t[32][33];
  int n = blockIdx.z;
  int s0 = blockIdx.x * 32, c0 = blockIdx.y * 32;
  int tx = threadIdx.x, ty = threadIdx.y;   // (32,8)
#pragma unroll
  for (int i = 0; i < 4; ++i) {
    int cl = ty + 8 * i;
    t[cl][tx] = x[((size_t)n * kC + c0 + cl) * kS + s0 + tx];
  }
  __syncthreads();
#pragma unroll
  for (int i = 0; i < 4; ++i) {
    int sl = ty + 8 * i;
    xT[((size_t)n * kS + s0 + sl) * kC + c0 + tx] = (__bf16)t[tx][sl];
  }
}

// ---------------- QKV GEMM: kqv = w_qkv @ x + b, route to q_tm/k_tm/v_dm ----------------
// 32x32 register block per wave (2 A-tiles x 2 B-tiles -> 4 WMMAs per k-step).
// A 32-row M-block spans exactly one head (dims 0..31).
// grid (18, 24, 4), block 128 (4 waves, wave = one 32x32 block, 8 K-steps of 32)
__global__ void qkv_kernel(const __bf16* __restrict__ xT, const __bf16* __restrict__ wq,
                           const float* __restrict__ bqkv,
                           __bf16* __restrict__ q_tm, __bf16* __restrict__ k_tm,
                           __bf16* __restrict__ v_dm) {
  int tid = threadIdx.x, wave = tid >> 5, lane = tid & 31;
  int nt2 = blockIdx.x * 4 + wave;         // 32-token block 0..71
  int mt2 = blockIdx.y;                    // 32-row block 0..23 (768 rows)
  int n   = blockIdx.z;
  int obase = mt2 * 32, tbase = nt2 * 32;
  int col = lane & 15, half = lane >> 4;

  v8f c00, c01, c10, c11;
  for (int i = 0; i < 8; ++i) { c00[i] = 0.f; c01[i] = 0.f; c10[i] = 0.f; c11[i] = 0.f; }

  const __bf16* xb = xT + (size_t)n * kS * kC + (size_t)tbase * kC;
  const __bf16* wb = wq + (size_t)obase * kC;
#pragma unroll
  for (int ks = 0; ks < 8; ++ks) {
    v16bf a0 = load_amat(wb + ks * 32, kC, lane);
    v16bf a1 = load_amat(wb + (size_t)16 * kC + ks * 32, kC, lane);
    v16bf b0 = load_bmat(xb + ks * 32, kC, lane);
    v16bf b1 = load_bmat(xb + (size_t)16 * kC + ks * 32, kC, lane);
    c00 = wmma_bf16(a0, b0, c00);
    c01 = wmma_bf16(a0, b1, c01);
    c10 = wmma_bf16(a1, b0, c10);
    c11 = wmma_bf16(a1, b1, c11);
  }

  int region = obase >> 8;                 // 0=K, 1=Q, 2=V
  int h = (obase & 255) >> 5;              // head (full head per 32-row block)
  size_t headoff = ((size_t)n * kNH + h) * kS;
  float bias[2][8];
#pragma unroll
  for (int mi = 0; mi < 2; ++mi)
#pragma unroll
    for (int r = 0; r < 8; ++r) bias[mi][r] = bqkv[obase + mi * 16 + 8 * half + r];

  if (region == 0) {                       // K -> token-major [n][h][tok][32]
#pragma unroll
    for (int mi = 0; mi < 2; ++mi)
#pragma unroll
      for (int ni = 0; ni < 2; ++ni) {
        const v8f& c = mi ? (ni ? c11 : c10) : (ni ? c01 : c00);
        int token = tbase + ni * 16 + col;
        __bf16* dst = k_tm + (headoff + token) * 32 + mi * 16 + 8 * half;
        v8bf pk;
#pragma unroll
        for (int r = 0; r < 8; ++r) pk[r] = (__bf16)(c[r] + bias[mi][r]);
        *(v8bf*)dst = pk;
      }
  } else if (region == 1) {                // Q (scaled) -> token-major
#pragma unroll
    for (int mi = 0; mi < 2; ++mi)
#pragma unroll
      for (int ni = 0; ni < 2; ++ni) {
        const v8f& c = mi ? (ni ? c11 : c10) : (ni ? c01 : c00);
        int token = tbase + ni * 16 + col;
        __bf16* dst = q_tm + (headoff + token) * 32 + mi * 16 + 8 * half;
        v8bf pk;
#pragma unroll
        for (int r = 0; r < 8; ++r)
          pk[r] = (__bf16)((c[r] + bias[mi][r]) * 0.0625f);  // DK^-0.5
        *(v8bf*)dst = pk;
      }
  } else {                                 // V -> dim-major [n][h][32][S]
#pragma unroll
    for (int mi = 0; mi < 2; ++mi)
#pragma unroll
      for (int ni = 0; ni < 2; ++ni) {
        const v8f& c = mi ? (ni ? c11 : c10) : (ni ? c01 : c00);
        int token = tbase + ni * 16 + col;
        __bf16* dst = v_dm + headoff * 32 + (size_t)(mi * 16 + 8 * half) * kS + token;
#pragma unroll
        for (int r = 0; r < 8; ++r) dst[(size_t)r * kS] = (__bf16)(c[r] + bias[mi][r]);
      }
  }
}

// ---------------- attention: per block = one (n,h,16-token row tile) ----------------
// grid (144, 8, 4), block 128, dynamic LDS ~225 KB (<= 320 KB WGP LDS)
__global__ void attn_kernel(const __bf16* __restrict__ q_tm, const __bf16* __restrict__ k_tm,
                            const __bf16* __restrict__ v_dm, __bf16* __restrict__ attn_tm) {
  extern __shared__ char smem[];
  float*  slog = (float*)smem;                                     // [16][kLPAD] f32 logits
  __bf16* sp   = (__bf16*)(smem + 16 * kLPAD * 4);                 // [16][kPPAD] bf16 probs
  float*  sacc = (float*)(smem + 16 * kLPAD * 4 + 16 * kPPAD * 2); // [16][32]
  float*  sred = sacc + 16 * 32;                                   // [16][8]
  float*  srmax = sred + 16 * 8;                                   // [16]
  float*  srsum = srmax + 16;                                      // [16]

  int tid = threadIdx.x, wave = tid >> 5, lane = tid & 31;
  int mt = blockIdx.x, h = blockIdx.y, n = blockIdx.z;
  int mbase = mt * 16;
  int col = lane & 15, half = lane >> 4;
  size_t headoff = ((size_t)n * kNH + h) * kS;

  for (int i = tid; i < 16 * 32; i += 128) sacc[i] = 0.f;

  // cached Q A-operand for this row tile (head_dim 32 == one WMMA K-step)
  v16bf aq = load_amat(q_tm + (headoff + mbase) * 32, 32, lane);

  // ---- logits = Q K^T, streamed into LDS ----
  const __bf16* kb_ = k_tm + headoff * 32;
  for (int nt = wave; nt < kNT; nt += 4) {
    v16bf b = load_bmat(kb_ + (size_t)nt * 16 * 32, 32, lane);
    v8f c; for (int i = 0; i < 8; ++i) c[i] = 0.f;
    c = wmma_bf16(aq, b, c);
    int nc = nt * 16 + col;
#pragma unroll
    for (int r = 0; r < 8; ++r) slog[(8 * half + r) * kLPAD + nc] = c[r];
  }
  __syncthreads();

  // ---- exact softmax in f32 (8 chunks of 288 per row) ----
  int row = tid & 15, chunk = tid >> 4;
  {
    float m = -3.4e38f;
    const float* lp = slog + row * kLPAD + chunk * 288;
    for (int i = 0; i < 288; ++i) m = fmaxf(m, lp[i]);
    sred[row * 8 + chunk] = m;
  }
  __syncthreads();
  if (tid < 16) {
    float m = sred[tid * 8];
    for (int i = 1; i < 8; ++i) m = fmaxf(m, sred[tid * 8 + i]);
    srmax[tid] = m;
  }
  __syncthreads();
  {
    float m = srmax[row], s = 0.f;
    const float* lp = slog + row * kLPAD + chunk * 288;
    __bf16* pp = sp + row * kPPAD + chunk * 288;
    for (int i = 0; i < 288; ++i) {
      float e = __expf(lp[i] - m);
      s += e;
      pp[i] = (__bf16)e;
    }
    sred[row * 8 + chunk] = s;
  }
  __syncthreads();
  if (tid < 16) {
    float s = 0.f;
    for (int i = 0; i < 8; ++i) s += sred[tid * 8 + i];
    srsum[tid] = s;
  }
  __syncthreads();

  // ---- O = P @ V : 72 K-steps of 32 tokens split over 4 waves ----
  const __bf16* vb = v_dm + headoff * 32;  // [n][h][32][S]
  v8f c0, c1;
  for (int i = 0; i < 8; ++i) { c0[i] = 0.f; c1[i] = 0.f; }
  for (int kt = wave; kt < 72; kt += 4) {
    v16bf a  = load_amat(sp + kt * 32, kPPAD, lane);                    // probs from LDS
    v16bf b0 = load_bmat(vb + (size_t)kt * 32, kS, lane);               // dims 0..15
    v16bf b1 = load_bmat(vb + (size_t)16 * kS + kt * 32, kS, lane);     // dims 16..31
    c0 = wmma_bf16(a, b0, c0);
    c1 = wmma_bf16(a, b1, c1);
  }
#pragma unroll
  for (int r = 0; r < 8; ++r) {
    atomicAdd(&sacc[(8 * half + r) * 32 + col], c0[r]);
    atomicAdd(&sacc[(8 * half + r) * 32 + col + 16], c1[r]);
  }
  __syncthreads();

  // ---- scale by 1/rowsum, emit token-major bf16 for projection B-operand ----
  {
    int j0 = tid * 4;                    // 512 elems / 128 threads
    int tok = j0 >> 5, d = j0 & 31;
    float inv = 1.f / srsum[tok];
    __bf16* dst = attn_tm + ((size_t)n * kS + mbase + tok) * 256 + h * 32 + d;
#pragma unroll
    for (int i = 0; i < 4; ++i) dst[i] = (__bf16)(sacc[tok * 32 + d + i] * inv);
  }
}

// ---------------- output projection: out = w_proj @ attn + b ----------------
// 32x32 register block per wave. grid (18, 8, 4), block 128
__global__ void proj_kernel(const __bf16* __restrict__ attn_tm, const __bf16* __restrict__ wp,
                            const float* __restrict__ bproj, float* __restrict__ out) {
  int tid = threadIdx.x, wave = tid >> 5, lane = tid & 31;
  int nt2 = blockIdx.x * 4 + wave;         // 32-token block 0..71
  int mt2 = blockIdx.y;                    // 32-row block 0..7 (256 rows)
  int n   = blockIdx.z;
  int obase = mt2 * 32, tbase = nt2 * 32;
  int col = lane & 15, half = lane >> 4;

  v8f c00, c01, c10, c11;
  for (int i = 0; i < 8; ++i) { c00[i] = 0.f; c01[i] = 0.f; c10[i] = 0.f; c11[i] = 0.f; }

  const __bf16* ab = attn_tm + (size_t)n * kS * 256 + (size_t)tbase * 256;
  const __bf16* wb = wp + (size_t)obase * 256;
#pragma unroll
  for (int ks = 0; ks < 8; ++ks) {
    v16bf a0 = load_amat(wb + ks * 32, 256, lane);
    v16bf a1 = load_amat(wb + (size_t)16 * 256 + ks * 32, 256, lane);
    v16bf b0 = load_bmat(ab + ks * 32, 256, lane);
    v16bf b1 = load_bmat(ab + (size_t)16 * 256 + ks * 32, 256, lane);
    c00 = wmma_bf16(a0, b0, c00);
    c01 = wmma_bf16(a0, b1, c01);
    c10 = wmma_bf16(a1, b0, c10);
    c11 = wmma_bf16(a1, b1, c11);
  }

#pragma unroll
  for (int mi = 0; mi < 2; ++mi)
#pragma unroll
    for (int ni = 0; ni < 2; ++ni) {
      const v8f& c = mi ? (ni ? c11 : c10) : (ni ? c01 : c00);
      int token = tbase + ni * 16 + col;
#pragma unroll
      for (int r = 0; r < 8; ++r) {
        int o = obase + mi * 16 + 8 * half + r;
        out[((size_t)n * 256 + o) * kS + token] = c[r] + bproj[o];
      }
    }
}

// ---------------- host ----------------
extern "C" void kernel_launch(void* const* d_in, const int* in_sizes, int n_in,
                              void* d_out, int out_size, void* d_ws, size_t ws_size,
                              hipStream_t stream) {
  const float* x      = (const float*)d_in[0];
  const float* w_qkv  = (const float*)d_in[1];
  const float* b_qkv  = (const float*)d_in[2];
  const float* w_proj = (const float*)d_in[3];
  const float* b_proj = (const float*)d_in[4];
  float* out = (float*)d_out;

  char* ws = (char*)d_ws;
  const size_t SZ_BIG = (size_t)kN * kS * 256 * 2;  // 4,718,592 B
  __bf16* xT      = (__bf16*)ws;               ws += SZ_BIG;
  __bf16* wq_bf   = (__bf16*)ws;               ws += (size_t)768 * 256 * 2;
  __bf16* wp_bf   = (__bf16*)ws;               ws += (size_t)256 * 256 * 2;
  __bf16* q_tm    = (__bf16*)ws;               ws += SZ_BIG;
  __bf16* k_tm    = (__bf16*)ws;               ws += SZ_BIG;
  __bf16* v_dm    = (__bf16*)ws;               ws += SZ_BIG;
  __bf16* attn_tm = (__bf16*)ws;               ws += SZ_BIG;

  wconv_kernel<<<768, 256, 0, stream>>>(w_qkv, w_proj, wq_bf, wp_bf);
  xpose_kernel<<<dim3(72, 8, 4), dim3(32, 8), 0, stream>>>(x, xT);
  qkv_kernel<<<dim3(18, 24, 4), 128, 0, stream>>>(xT, wq_bf, b_qkv, q_tm, k_tm, v_dm);

  size_t smem = (size_t)16 * kLPAD * 4 + (size_t)16 * kPPAD * 2 +
                (size_t)(16 * 32 + 16 * 8 + 16 + 16) * 4;  // 224,896 B
  attn_kernel<<<dim3(144, 8, 4), 128, smem, stream>>>(q_tm, k_tm, v_dm, attn_tm);

  proj_kernel<<<dim3(18, 8, 4), 128, 0, stream>>>(attn_tm, wp_bf, b_proj, out);
}
